// HandTornRNN_86681029968353
// MI455X (gfx1250) — matrix operand order repaired and checked by reading, not compiled, not run on hardware
//
#include <hip/hip_runtime.h>
#include <hip/hip_bf16.h>
#include <math.h>

typedef __bf16 bf16_t;
typedef __attribute__((ext_vector_type(16))) __bf16 v16bf;
typedef __attribute__((ext_vector_type(8)))  __bf16 v8bf;
typedef __attribute__((ext_vector_type(8)))  float   v8f;

#define RNN_I   1024
#define RNN_H   1024
#define RNN_B   32
#define RNN_S   1024
#define APITCH  1032   // bf16 elems; 516 dwords/row -> 4-bank shift/row, conflict-free b128 LDS reads

// ---------------------------------------------------------------------------
// f32 -> bf16 elementwise convert, 8 elements/thread (memory-bound pass)
// ---------------------------------------------------------------------------
__global__ void cvt_f32_to_bf16(const float* __restrict__ in,
                                bf16_t* __restrict__ out, int n8) {
  int i = blockIdx.x * blockDim.x + threadIdx.x;
  if (i >= n8) return;
  const float* p = in + (long)i * 8;
  v8bf v;
#pragma unroll
  for (int j = 0; j < 8; ++j) v[j] = (bf16_t)p[j];
  *(v8bf*)(out + (long)i * 8) = v;
}

// ---------------------------------------------------------------------------
// 1024x1024 f32 -> bf16 transpose via LDS (W[k][n] -> WT[n][k], k-contiguous)
// ---------------------------------------------------------------------------
__global__ void transpose_to_bf16(const float* __restrict__ W,
                                  bf16_t* __restrict__ WT) {
  __shared__ bf16_t tile[32][33];
  int x = blockIdx.x * 32 + threadIdx.x;  // col n
  int y = blockIdx.y * 32 + threadIdx.y;  // row k
  tile[threadIdx.y][threadIdx.x] = (bf16_t)W[y * RNN_H + x];
  __syncthreads();
  int n = blockIdx.x * 32 + threadIdx.y;
  int k = blockIdx.y * 32 + threadIdx.x;
  WT[n * RNN_I + k] = tile[threadIdx.x][threadIdx.y];
}

__global__ void zero_bf16(bf16_t* __restrict__ p, int n) {
  int i = blockIdx.x * blockDim.x + threadIdx.x;
  if (i < n) p[i] = (bf16_t)0.0f;
}

// ---------------------------------------------------------------------------
// GEMM1: Xp[32768,1024] = Xbf @ Wxh + b  (bf16 WMMA, f32 accumulate)
// Block = 8 waves sharing one 16-row A tile staged in LDS; each wave computes
// a 16x64 strip (4 N-tiles). B double-buffered in registers so global loads
// for k+32 overlap the WMMAs at k. grid 4096 x 256.
// ---------------------------------------------------------------------------
__global__ void __launch_bounds__(256)
xproj_gemm(const bf16_t* __restrict__ Xbf, const bf16_t* __restrict__ WxhT,
           const float* __restrict__ bias, float* __restrict__ Xp) {
  __shared__ bf16_t As[16 * APITCH];

  const int lane = threadIdx.x & 31;
  const int wave = threadIdx.x >> 5;
  const int gw   = blockIdx.x * 8 + wave;
  const int tm   = gw >> 4;            // constant within a block
  const int n0   = (gw & 15) * 64;     // 4 N-tiles of 16
  const int l15  = lane & 15;
  const int grp  = lane >> 4;

  // Stage A tile: 16 rows x 1024 bf16 (32 KB) -> LDS, 256 threads x 8 chunks
  const bf16_t* Ag = Xbf + (long)tm * 16 * RNN_I;
#pragma unroll
  for (int i = 0; i < 8; ++i) {
    int c    = i * 256 + threadIdx.x;        // 0..2047 chunks of 8 elems
    int row  = c >> 7;
    int col8 = (c & 127) * 8;
    *(v8bf*)(&As[row * APITCH + col8]) = *(const v8bf*)(Ag + row * RNN_I + col8);
  }

  v8f acc[4];
#pragma unroll
  for (int j = 0; j < 4; ++j) {
    float bv = bias[n0 + j * 16 + l15];      // C[m][n] = b[n] for all m
#pragma unroll
    for (int r = 0; r < 8; ++r) acc[j][r] = bv;
  }
  __syncthreads();

  // B double buffer (prefetch one k-chunk ahead; the k=1024 over-read of the
  // last row lands in the adjacent workspace region and is never consumed).
  v16bf bcur[4], bnxt[4];
#pragma unroll
  for (int j = 0; j < 4; ++j)
    bcur[j] = *(const v16bf*)(WxhT + (long)(n0 + j * 16 + l15) * RNN_I +
                              grp * 16);

  for (int k0 = 0; k0 < RNN_I; k0 += 32) {
#pragma unroll
    for (int j = 0; j < 4; ++j)
      bnxt[j] = *(const v16bf*)(WxhT + (long)(n0 + j * 16 + l15) * RNN_I +
                                (k0 + 32) + grp * 16);
    // A operand from LDS: lane m=l15; v0-7 = K[grp*16..+16)
    v8bf alo = *(const v8bf*)(&As[l15 * APITCH + k0 + grp * 16]);
    v8bf ahi = *(const v8bf*)(&As[l15 * APITCH + k0 + grp * 16 + 8]);
    v16bf a;
#pragma unroll
    for (int i = 0; i < 8; ++i) { a[i] = alo[i]; a[i + 8] = ahi[i]; }
#pragma unroll
    for (int j = 0; j < 4; ++j)
      acc[j] = __builtin_amdgcn_wmma_f32_16x16x32_bf16(
          false, a, false, bcur[j], (short)0, acc[j], false, false);
#pragma unroll
    for (int j = 0; j < 4; ++j) bcur[j] = bnxt[j];
  }

#pragma unroll
  for (int j = 0; j < 4; ++j) {
    int col = n0 + j * 16 + l15;
#pragma unroll
    for (int r = 0; r < 8; ++r) {
      int row = tm * 16 + r + grp * 8;       // C layout: M = r + grp*8
      Xp[(long)row * RNN_H + col] = acc[j][r];
    }
  }
}

// ---------------------------------------------------------------------------
// One recurrent step: Hout = tanh(Xp[:,t,:] + Hin @ Whh), writes f32 out.
// 128 tiles (2 M x 64 N); one wave per tile; 8 waves/block share one 16-row
// H tile staged in LDS. B (Whh) double-buffered in registers. grid 16 x 256.
// Whh (2MB bf16) and H (64KB) stay resident in the 192MB L2 across all steps.
// ---------------------------------------------------------------------------
__global__ void __launch_bounds__(256)
rnn_step(const bf16_t* __restrict__ Hin, bf16_t* __restrict__ Hout,
         const bf16_t* __restrict__ WhhT, const float* __restrict__ Xp,
         float* __restrict__ out, int t) {
  __shared__ bf16_t Hs[16 * APITCH];

  const int lane = threadIdx.x & 31;
  const int wave = threadIdx.x >> 5;
  const int mt   = blockIdx.x >> 3;                 // 0..1, shared by block
  const int nt   = (blockIdx.x & 7) * 8 + wave;     // 0..63
  const int n0   = nt * 16;
  const int l15  = lane & 15;
  const int grp  = lane >> 4;
  const int n    = n0 + l15;

  // Stage H tile (16 batch rows x 1024 bf16) into LDS
  const bf16_t* Hg = Hin + (long)mt * 16 * RNN_H;
#pragma unroll
  for (int i = 0; i < 8; ++i) {
    int c    = i * 256 + threadIdx.x;
    int row  = c >> 7;
    int col8 = (c & 127) * 8;
    *(v8bf*)(&Hs[row * APITCH + col8]) = *(const v8bf*)(Hg + row * RNN_H + col8);
  }

  v8f acc;
#pragma unroll
  for (int r = 0; r < 8; ++r) {
    int b = mt * 16 + r + grp * 8;                  // batch index
    acc[r] = Xp[((long)b * RNN_S + t) * RNN_H + n];
  }
  __syncthreads();

  const bf16_t* bcol = WhhT + (long)n * RNN_H;
  v16bf bcur = *(const v16bf*)(bcol + grp * 16);
  for (int k0 = 0; k0 < RNN_H; k0 += 32) {
    v16bf bnxt = *(const v16bf*)(bcol + (k0 + 32) + grp * 16);
    v8bf alo = *(const v8bf*)(&Hs[l15 * APITCH + k0 + grp * 16]);
    v8bf ahi = *(const v8bf*)(&Hs[l15 * APITCH + k0 + grp * 16 + 8]);
    v16bf a;
#pragma unroll
    for (int i = 0; i < 8; ++i) { a[i] = alo[i]; a[i + 8] = ahi[i]; }
    acc = __builtin_amdgcn_wmma_f32_16x16x32_bf16(
        false, a, false, bcur, (short)0, acc, false, false);
    bcur = bnxt;
  }

#pragma unroll
  for (int r = 0; r < 8; ++r) {
    int b = mt * 16 + r + grp * 8;
    float h = tanhf(acc[r]);
    out[((long)b * RNN_S + t) * RNN_H + n] = h;     // out[b, t, n]
    Hout[(long)b * RNN_H + n] = (bf16_t)h;          // next step's A matrix
    if (t == RNN_S - 1)                             // H_last tail of d_out
      out[(long)RNN_B * RNN_S * RNN_H + (long)b * RNN_H + n] = h;
  }
}

// ---------------------------------------------------------------------------
extern "C" void kernel_launch(void* const* d_in, const int* in_sizes, int n_in,
                              void* d_out, int out_size, void* d_ws,
                              size_t ws_size, hipStream_t stream) {
  (void)in_sizes; (void)n_in; (void)out_size; (void)ws_size;
  const float* X   = (const float*)d_in[0];
  const float* Wxh = (const float*)d_in[1];
  const float* Whh = (const float*)d_in[2];
  const float* bh  = (const float*)d_in[3];
  float* out = (float*)d_out;

  // Workspace layout (~200 MB):
  char* ws = (char*)d_ws;
  bf16_t* WxhT = (bf16_t*)(ws);                               //   2 MB
  bf16_t* WhhT = (bf16_t*)(ws + (size_t)(2u  << 20));         //   2 MB
  bf16_t* Hb0  = (bf16_t*)(ws + (size_t)(4u  << 20));         //  64 KB
  bf16_t* Hb1  = (bf16_t*)(ws + (size_t)(4u  << 20) + (1u << 16)); // 64 KB
  bf16_t* Xbf  = (bf16_t*)(ws + (size_t)(8u  << 20));         //  64 MB
  float*  Xp   = (float*) (ws + (size_t)(72u << 20));         // 128 MB

  const int nX8 = RNN_B * RNN_S * RNN_I / 8;  // 4,194,304
  cvt_f32_to_bf16<<<(nX8 + 255) / 256, 256, 0, stream>>>(X, Xbf, nX8);

  dim3 tb(32, 32), tg(32, 32);
  transpose_to_bf16<<<tg, tb, 0, stream>>>(Wxh, WxhT);
  transpose_to_bf16<<<tg, tb, 0, stream>>>(Whh, WhhT);
  zero_bf16<<<(RNN_B * RNN_H + 255) / 256, 256, 0, stream>>>(Hb0,
                                                             RNN_B * RNN_H);

  // Xp = X @ W_xh + b : 2048 M-tiles * 16 strips = 32768 waves
  xproj_gemm<<<4096, 256, 0, stream>>>(Xbf, WxhT, bh, Xp);

  // Sequential scan; one graph node per timestep gives device-wide ordering.
  for (int t = 0; t < RNN_S; ++t) {
    const bf16_t* Hi = (t & 1) ? Hb1 : Hb0;
    bf16_t*       Ho = (t & 1) ? Hb0 : Hb1;
    rnn_step<<<16, 256, 0, stream>>>(Hi, Ho, WhhT, Xp, out, t);
  }
}
